// HierCommAgent_52355651338840
// MI455X (gfx1250) — compile-verified, matrix operand order checked by reading, not run on hardware
//
#include <hip/hip_runtime.h>
#include <hip/hip_bf16.h>
#include <math.h>

// ---------------- problem constants ----------------
#define NAGENTS 2048
#define HIDD    512
#define OBSD    512
#define NHEADS  8
#define HDIM    64      // 512 / 8
#define NACT    128

// ---------------- WMMA tile config ----------------
#define BM  128
#define BN  128
#define BK  32
#define LDT 40          // bf16 LDS row stride (80B, 16B aligned)  [manual kernel]
#define PK  36          // fp32 LDS row stride (144B, 16B aligned) [TDM kernel]

typedef __attribute__((ext_vector_type(16))) __bf16 v16bf;
typedef __attribute__((ext_vector_type(8)))  __bf16 v8bf;
typedef __attribute__((ext_vector_type(8)))  float  v8f;
typedef __attribute__((ext_vector_type(4)))  unsigned v4u;
typedef __attribute__((ext_vector_type(8)))  int      v8i;
typedef __attribute__((ext_vector_type(4)))  int      v4i;

#if defined(__gfx1250__) && __has_builtin(__builtin_amdgcn_tensor_load_to_lds)
#define USE_TDM 1
#endif

// Fragment load from bf16 LDS tile: lane (l&15) = row, half=(l>>4) selects k-subrange
// CDNA5 16-bit A/B operand layout: half 0 -> k {0..7,16..23}, half 1 -> k {8..15,24..31}
__device__ __forceinline__ v16bf frag_ld(const __bf16* p) {
  v8bf c0 = *(const v8bf*)(p);
  v8bf c1 = *(const v8bf*)(p + 16);
  v16bf f;
#pragma unroll
  for (int i = 0; i < 8; ++i) { f[i] = c0[i]; f[i + 8] = c1[i]; }
  return f;
}

// Fragment load from fp32 LDS tile + convert to bf16 (v_cvt_pk_bf16_f32)
__device__ __forceinline__ v16bf frag_ld_f32(const float* p) {
  float4 a0 = *(const float4*)(p);
  float4 a1 = *(const float4*)(p + 4);
  float4 b0 = *(const float4*)(p + 16);
  float4 b1 = *(const float4*)(p + 20);
  v16bf f;
  f[0]=(__bf16)a0.x;  f[1]=(__bf16)a0.y;  f[2]=(__bf16)a0.z;  f[3]=(__bf16)a0.w;
  f[4]=(__bf16)a1.x;  f[5]=(__bf16)a1.y;  f[6]=(__bf16)a1.z;  f[7]=(__bf16)a1.w;
  f[8]=(__bf16)b0.x;  f[9]=(__bf16)b0.y;  f[10]=(__bf16)b0.z; f[11]=(__bf16)b0.w;
  f[12]=(__bf16)b1.x; f[13]=(__bf16)b1.y; f[14]=(__bf16)b1.z; f[15]=(__bf16)b1.w;
  return f;
}

#ifdef USE_TDM
// Issue one TDM 2D tile load: 32 fp32 x 128 rows, row stride ldElems, into LDS at
// byte offset ldsOff, padded +4 dwords every 32 dwords (-> 36-dword LDS row stride).
// Rows beyond tensorH are zero-filled by the TDM (replaces bounds guards).
// D# packing per CDNA5 ISA 8.3 (group0) / 8.4 (group1). 6-arg builtin form
// (amdgpu-toolchain / clang-23): (v4u, v8i, v4i, v4i, v8i, i32 cpol).
__device__ __forceinline__ void tdm_load_2d(unsigned ldsOff, const float* gptr,
                                            int tensorH, int ldElems)
{
  unsigned long long ga = (unsigned long long)(uintptr_t)gptr;
  v4u g0;
  g0[0] = 1u;                                             // count=1, user descriptor
  g0[1] = ldsOff;                                         // lds_addr (bytes)
  g0[2] = (unsigned)ga;                                   // global_addr[31:0]
  g0[3] = (unsigned)((ga >> 32) & 0x01ffffffu) | (2u << 30); // addr[56:32] | type=2
  v8i g1;
  unsigned d0 = (2u << 16)      // data_size = 4B
              | (1u << 20)      // pad_enable
              | (4u << 22)      // pad_interval: 32 dwords
              | (3u << 25);     // pad_amount: 4 dwords
  g1[0] = (int)d0;
  g1[1] = (int)(32u << 16);                               // tensor_dim0[15:0] = 32
  g1[2] = (int)(((unsigned)tensorH & 0xffffu) << 16);     // dim0[31:16]=0 | dim1[15:0]
  g1[3] = (int)((((unsigned)tensorH >> 16) & 0xffffu) | (32u << 16)); // dim1 hi | tile_dim0=32
  g1[4] = (int)128u;                                      // tile_dim1=128, tile_dim2=0
  g1[5] = (int)(unsigned)ldElems;                         // tensor_dim0_stride[31:0]
  g1[6] = 0;                                              // stride[47:32]=0, dim1_stride lo=0
  g1[7] = 0;
  v4i z4; z4[0] = z4[1] = z4[2] = z4[3] = 0;
  v8i z8;
#pragma unroll
  for (int i = 0; i < 8; ++i) z8[i] = 0;
  __builtin_amdgcn_tensor_load_to_lds(g0, g1, z4, z4, z8, 0);
}
#endif

// ============================================================================
// TDM-staged GEMM: Y[M,Nt] = act( X[M,K] @ B^T + bias ), B is [Nt,K] row-major.
// fp32 LDS tiles staged by the Tensor Data Mover; bf16 WMMA, fp32 accumulate.
// Requires: M % 128 == 0, K % 32 == 0, 16B-aligned K slices. Nt arbitrary.
// ============================================================================
__global__ __launch_bounds__(256)
void wmma_gemm_tdm(const float* __restrict__ X, int ldx,
                   const float* __restrict__ B, int ldb,
                   const float* __restrict__ bias,
                   float* __restrict__ Y, int ldy,
                   int M, int Nt, int K, int actTanh)
{
  __shared__ __align__(16) float As[BM * PK];
  __shared__ __align__(16) float Bs[BN * PK];

  const int tid  = threadIdx.x;
  const int m0   = blockIdx.y * BM;
  const int n0   = blockIdx.x * BN;
  const int lane = tid & 31;
  const int wave = tid >> 5;
  const int wm   = (wave >> 1) * 32;
  const int wn   = (wave & 1) * 64;
  const int half = lane >> 4;
  const int row  = lane & 15;

  v8f acc[2][4];
#pragma unroll
  for (int mi = 0; mi < 2; ++mi)
#pragma unroll
    for (int ni = 0; ni < 4; ++ni)
#pragma unroll
      for (int j = 0; j < 8; ++j) acc[mi][ni][j] = 0.0f;

  for (int k0 = 0; k0 < K; k0 += BK) {
#ifdef USE_TDM
    if (wave == 0) {
      tdm_load_2d((unsigned)(uintptr_t)(void*)As, X + (size_t)m0 * ldx + k0, M - m0, ldx);
      tdm_load_2d((unsigned)(uintptr_t)(void*)Bs, B + (size_t)n0 * ldb + k0, Nt - n0, ldb);
      __builtin_amdgcn_s_wait_tensorcnt(0);
    }
#else
    // manual staging fallback (host pass / toolchains without the TDM builtin)
#pragma unroll
    for (int it = 0; it < 4; ++it) {
      int idx = (tid + it * 256) << 2;
      int r   = idx >> 5;
      int kk  = idx & 31;
      float4 va = *(const float4*)(X + (size_t)(m0 + r) * ldx + (k0 + kk));
      *(float4*)&As[r * PK + kk] = va;
      float4 vb = make_float4(0.f, 0.f, 0.f, 0.f);
      if (n0 + r < Nt) vb = *(const float4*)(B + (size_t)(n0 + r) * ldb + (k0 + kk));
      *(float4*)&Bs[r * PK + kk] = vb;
    }
#endif
    __syncthreads();

    v16bf af[2], bfrag[4];
#pragma unroll
    for (int mi = 0; mi < 2; ++mi)
      af[mi] = frag_ld_f32(&As[(wm + mi * 16 + row) * PK + half * 8]);
#pragma unroll
    for (int ni = 0; ni < 4; ++ni)
      bfrag[ni] = frag_ld_f32(&Bs[(wn + ni * 16 + row) * PK + half * 8]);

#pragma unroll
    for (int mi = 0; mi < 2; ++mi)
#pragma unroll
      for (int ni = 0; ni < 4; ++ni)
        acc[mi][ni] = __builtin_amdgcn_wmma_f32_16x16x32_bf16(
            false, af[mi], false, bfrag[ni], (short)0, acc[mi][ni], false, false);

    __syncthreads();
  }

#pragma unroll
  for (int mi = 0; mi < 2; ++mi) {
#pragma unroll
    for (int ni = 0; ni < 4; ++ni) {
      int cc = n0 + wn + ni * 16 + (lane & 15);
      if (cc < Nt) {
        float bv = bias ? bias[cc] : 0.0f;
#pragma unroll
        for (int j = 0; j < 8; ++j) {
          int rr = m0 + wm + mi * 16 + (lane >> 4) * 8 + j;
          float v = acc[mi][ni][j] + bv;
          if (actTanh) v = tanhf(v);
          Y[(size_t)rr * ldy + cc] = v;
        }
      }
    }
  }
}

// ============================================================================
// Manual-staged GEMM (bf16 LDS tiles) — used for the transposed-B case (P @ V):
//   transB==1 : B is [K,Nt] row-major, op(B)=B   (transposed into LDS at stage)
//   transB==0 : B is [Nt,K] row-major, op(B)=B^T (kept for completeness)
// ============================================================================
__global__ __launch_bounds__(256)
void wmma_gemm(const float* __restrict__ X, int ldx,
               const float* __restrict__ B, int ldb,
               const float* __restrict__ bias,
               float* __restrict__ Y, int ldy,
               int M, int Nt, int K, int transB, int actTanh)
{
  __shared__ __align__(16) __bf16 As[BM * LDT];
  __shared__ __align__(16) __bf16 Bs[BN * LDT];

  const int tid  = threadIdx.x;
  const int m0   = blockIdx.y * BM;
  const int n0   = blockIdx.x * BN;
  const int lane = tid & 31;
  const int wave = tid >> 5;
  const int wm   = (wave >> 1) * 32;
  const int wn   = (wave & 1) * 64;
  const int half = lane >> 4;
  const int row  = lane & 15;

  v8f acc[2][4];
#pragma unroll
  for (int mi = 0; mi < 2; ++mi)
#pragma unroll
    for (int ni = 0; ni < 4; ++ni)
#pragma unroll
      for (int j = 0; j < 8; ++j) acc[mi][ni][j] = 0.0f;

  for (int k0 = 0; k0 < K; k0 += BK) {
#pragma unroll
    for (int it = 0; it < 4; ++it) {
      int idx = (tid + it * 256) << 2;
      int r   = idx >> 5;
      int kk  = idx & 31;
      float4 v = *(const float4*)(X + (size_t)(m0 + r) * ldx + (k0 + kk));
      __bf16* d = &As[r * LDT + kk];
      d[0] = (__bf16)v.x; d[1] = (__bf16)v.y; d[2] = (__bf16)v.z; d[3] = (__bf16)v.w;
    }
    if (!transB) {
#pragma unroll
      for (int it = 0; it < 4; ++it) {
        int idx = (tid + it * 256) << 2;
        int r   = idx >> 5;
        int kk  = idx & 31;
        float4 v = make_float4(0.f, 0.f, 0.f, 0.f);
        if (n0 + r < Nt) v = *(const float4*)(B + (size_t)(n0 + r) * ldb + (k0 + kk));
        __bf16* d = &Bs[r * LDT + kk];
        d[0] = (__bf16)v.x; d[1] = (__bf16)v.y; d[2] = (__bf16)v.z; d[3] = (__bf16)v.w;
      }
    } else {
#pragma unroll
      for (int it = 0; it < 16; ++it) {
        int idx = tid + it * 256;
        int kk  = idx >> 7;
        int nn  = idx & 127;
        float v = (n0 + nn < Nt) ? B[(size_t)(k0 + kk) * ldb + (n0 + nn)] : 0.0f;
        Bs[nn * LDT + kk] = (__bf16)v;
      }
    }
    if (k0 + BK < K)
      __builtin_prefetch(X + (size_t)(m0 + (tid >> 1)) * ldx + (k0 + BK), 0, 0);

    __syncthreads();

    v16bf af[2], bfrag[4];
#pragma unroll
    for (int mi = 0; mi < 2; ++mi)
      af[mi] = frag_ld(&As[(wm + mi * 16 + row) * LDT + half * 8]);
#pragma unroll
    for (int ni = 0; ni < 4; ++ni)
      bfrag[ni] = frag_ld(&Bs[(wn + ni * 16 + row) * LDT + half * 8]);

#pragma unroll
    for (int mi = 0; mi < 2; ++mi)
#pragma unroll
      for (int ni = 0; ni < 4; ++ni)
        acc[mi][ni] = __builtin_amdgcn_wmma_f32_16x16x32_bf16(
            false, af[mi], false, bfrag[ni], (short)0, acc[mi][ni], false, false);

    __syncthreads();
  }

#pragma unroll
  for (int mi = 0; mi < 2; ++mi) {
#pragma unroll
    for (int ni = 0; ni < 4; ++ni) {
      int cc = n0 + wn + ni * 16 + (lane & 15);
      if (cc < Nt) {
        float bv = bias ? bias[cc] : 0.0f;
#pragma unroll
        for (int j = 0; j < 8; ++j) {
          int rr = m0 + wm + mi * 16 + (lane >> 4) * 8 + j;
          float v = acc[mi][ni][j] + bv;
          if (actTanh) v = tanhf(v);
          Y[(size_t)rr * ldy + cc] = v;
        }
      }
    }
  }
}

// Row softmax, one block per row; in-place safe.
// mode 0: softmax | 1: mask cid[col]==cid[row] | 2: mask counts[col]>0 | 3: log_softmax
__global__ __launch_bounds__(256)
void softmax_rows(float* __restrict__ Y, const float* __restrict__ Xin,
                  int ld, int ncols, float scale, int mode,
                  const int* __restrict__ cid, const int* __restrict__ counts)
{
  __shared__ float red[256];
  const int r   = blockIdx.x;
  const int tid = threadIdx.x;
  const int myc = (mode == 1) ? cid[r] : 0;

  float mx = -3.0e38f;
  for (int j = tid; j < ncols; j += 256) {
    float s = Xin[(size_t)r * ld + j] * scale;
    if (mode == 1 && cid[j] != myc)   s = -1.0e9f;
    if (mode == 2 && counts[j] <= 0)  s = -1.0e9f;
    mx = fmaxf(mx, s);
  }
  red[tid] = mx; __syncthreads();
  for (int o = 128; o > 0; o >>= 1) { if (tid < o) red[tid] = fmaxf(red[tid], red[tid + o]); __syncthreads(); }
  mx = red[0]; __syncthreads();

  float sum = 0.0f;
  for (int j = tid; j < ncols; j += 256) {
    float s = Xin[(size_t)r * ld + j] * scale;
    if (mode == 1 && cid[j] != myc)   s = -1.0e9f;
    if (mode == 2 && counts[j] <= 0)  s = -1.0e9f;
    sum += __expf(s - mx);
  }
  red[tid] = sum; __syncthreads();
  for (int o = 128; o > 0; o >>= 1) { if (tid < o) red[tid] += red[tid + o]; __syncthreads(); }
  sum = red[0]; __syncthreads();

  const float inv = 1.0f / sum;
  const float lse = __logf(sum);
  for (int j = tid; j < ncols; j += 256) {
    float s = Xin[(size_t)r * ld + j] * scale;
    if (mode == 1 && cid[j] != myc)   s = -1.0e9f;
    if (mode == 2 && counts[j] <= 0)  s = -1.0e9f;
    Y[(size_t)r * ld + j] = (mode == 3) ? (s - mx - lse) : __expf(s - mx) * inv;
  }
}

// ---------------- small helper kernels ----------------
__global__ void fzero(float* p, int n) {
  int i = blockIdx.x * blockDim.x + threadIdx.x;
  if (i < n) p[i] = 0.0f;
}

__global__ void concat2(const float* __restrict__ A, const float* __restrict__ Bm,
                        float* __restrict__ Y, int n, int w) {
  int idx = blockIdx.x * blockDim.x + threadIdx.x;
  int tot = 2 * w;
  if (idx >= n * tot) return;
  int i = idx / tot, c = idx % tot;
  Y[idx] = (c < w) ? A[(size_t)i * w + c] : Bm[(size_t)i * w + (c - w)];
}

__device__ __forceinline__ unsigned fkey(float f) {
  unsigned u = __float_as_uint(f);
  return (u & 0x80000000u) ? ~u : (u | 0x80000000u);
}
__device__ __forceinline__ float keyf(unsigned k) {
  unsigned u = (k & 0x80000000u) ? (k & 0x7fffffffu) : ~k;
  return __uint_as_float(u);
}

__global__ void seg_init(int* counts, unsigned* mkey, float* wsum, int n) {
  int i = blockIdx.x * blockDim.x + threadIdx.x;
  if (i < n) { counts[i] = 0; mkey[i] = 0u; wsum[i] = 0.0f; }
}
__global__ void seg_pass1(const int* cid, const float* sc, int* counts, unsigned* mkey, int n) {
  int i = blockIdx.x * blockDim.x + threadIdx.x;
  if (i < n) {
    atomicAdd(&counts[cid[i]], 1);
    atomicMax(&mkey[cid[i]], fkey(sc[i]));
  }
}
__global__ void seg_pass2(const int* cid, const float* sc, const unsigned* mkey,
                          float* wv, float* wsum, int n) {
  int i = blockIdx.x * blockDim.x + threadIdx.x;
  if (i < n) {
    float m = keyf(mkey[cid[i]]);
    float w = __expf(sc[i] - m);
    wv[i] = w;
    atomicAdd(&wsum[cid[i]], w);
  }
}
__global__ void reps_accum(const int* cid, const float* wv, const float* local,
                           float* reps, int n, int h) {
  int idx = blockIdx.x * blockDim.x + threadIdx.x;
  if (idx < n * h) {
    int i = idx / h, f = idx % h;
    atomicAdd(&reps[(size_t)cid[i] * h + f], wv[i] * local[idx]);
  }
}
__global__ void reps_norm(float* reps, const float* wsum, int n, int h) {
  int idx = blockIdx.x * blockDim.x + threadIdx.x;
  if (idx < n * h) {
    float s = wsum[idx / h];
    reps[idx] = (s > 0.0f) ? reps[idx] / s : 0.0f;
  }
}
__global__ void build_after(const float* __restrict__ local, const float* __restrict__ interA,
                            const float* __restrict__ intra, const int* __restrict__ cid,
                            const int* __restrict__ counts, float* __restrict__ after,
                            int n, int h) {
  int idx = blockIdx.x * blockDim.x + threadIdx.x;
  if (idx < n * h) {
    int i = idx / h, f = idx % h;
    int c = cid[i];
    float lv = local[idx];
    size_t ro = (size_t)i * (3 * h);
    after[ro + f]         = lv;
    after[ro + h + f]     = interA[(size_t)c * h + f];
    after[ro + 2 * h + f] = (counts[c] > 1) ? intra[idx] : lv;
  }
}

// ---------------- host-side pipeline ----------------
extern "C" void kernel_launch(void* const* d_in, const int* in_sizes, int n_in,
                              void* d_out, int out_size, void* d_ws, size_t ws_size,
                              hipStream_t stream)
{
  (void)in_sizes; (void)n_in; (void)out_size; (void)ws_size;
  const int N = NAGENTS, H = HIDD, O = OBSD, D = HDIM, A = NACT;
  const int H3 = 3 * H;

  const float* obs      = (const float*)d_in[0];
  const float* c_fc1_w  = (const float*)d_in[1];  const float* c_fc1_b = (const float*)d_in[2];
  const float* c_wi     = (const float*)d_in[3];  const float* c_bi    = (const float*)d_in[4];
  const float* c_wo     = (const float*)d_in[5];  const float* c_bo    = (const float*)d_in[6];
  const float* c_fc2_w  = (const float*)d_in[7];  const float* c_fc2_b = (const float*)d_in[8];
  const float* c_head_w = (const float*)d_in[9];  const float* c_head_b= (const float*)d_in[10];
  const float* t_local_w= (const float*)d_in[11]; const float* t_local_b=(const float*)d_in[12];
  const float* ia_wi    = (const float*)d_in[13]; const float* ia_bi   = (const float*)d_in[14];
  const float* ia_wo    = (const float*)d_in[15]; const float* ia_bo   = (const float*)d_in[16];
  const float* ie_wi    = (const float*)d_in[17]; const float* ie_bi   = (const float*)d_in[18];
  const float* ie_wo    = (const float*)d_in[19]; const float* ie_bo   = (const float*)d_in[20];
  const float* set_w    = (const float*)d_in[21]; const float* set_b   = (const float*)d_in[22];
  const float* a_fc1_w  = (const float*)d_in[23]; const float* a_fc1_b = (const float*)d_in[24];
  const float* a_fc2_w  = (const float*)d_in[25]; const float* a_fc2_b = (const float*)d_in[26];
  const float* actor_w  = (const float*)d_in[27]; const float* actor_b = (const float*)d_in[28];
  const float* value_w  = (const float*)d_in[29]; const float* value_b = (const float*)d_in[30];
  const int*   cid      = (const int*)  d_in[31];

  // workspace layout (floats); heavy reuse. Total = 17*N*H + 5*N floats ~= 68 MB
  float* w      = (float*)d_ws;
  float* x1     = w;                          // N*H  (later: reps)
  float* qkv    = x1     + (size_t)N * H;     // N*3H
  float* scores = qkv    + (size_t)N * H3;    // N*N (per-head, reused)
  float* attno  = scores + (size_t)N * N;     // N*H
  float* hbuf   = attno  + (size_t)N * H;     // N*H  (later: intra)
  float* cat2   = hbuf   + (size_t)N * H;     // N*2H (later: ah1 | ah2)
  float* zbuf   = cat2   + (size_t)2 * N * H; // N*H  (later: inter_all)
  float* local  = zbuf   + (size_t)N * H;     // N*H
  float* after  = local  + (size_t)N * H;     // N*3H
  float* sc     = after  + (size_t)N * H3;    // N
  float* wv     = sc + N;                     // N
  float* wsum   = wv + N;                     // N
  int*      counts = (int*)(wsum + N);        // N
  unsigned* mkey   = (unsigned*)(counts + N); // N

  float* cmat = (float*)d_out;                // N*N
  float* alog = cmat + (size_t)N * N;         // N*A
  float* vout = alog + (size_t)N * A;         // N

  dim3 blk(256);
  auto grd = [](int M, int Nt) {
    return dim3((unsigned)((Nt + BN - 1) / BN), (unsigned)((M + BM - 1) / BM));
  };

  // Y = act(X @ W^T + b): TDM-staged WMMA GEMM
  auto gemmW = [&](const float* Xp, int ldx, const float* Wp, int ldw, const float* bp,
                   float* Yp, int ldy, int M, int Nt, int K, int act) {
    wmma_gemm_tdm<<<grd(M, Nt), blk, 0, stream>>>(Xp, ldx, Wp, ldw, bp, Yp, ldy, M, Nt, K, act);
  };

  // one full multi-head attention block (per-head decomposed)
  auto mha = [&](const float* xin, const float* wi, const float* bi,
                 const float* wo, const float* bo, float* yout, int maskMode) {
    gemmW(xin, H, wi, H, bi, qkv, H3, N, H3, H, 0);
    for (int h = 0; h < NHEADS; ++h) {
      // S = Q_h @ K_h^T
      gemmW(qkv + h * D, H3, qkv + H + h * D, H3, nullptr, scores, N, N, N, D, 0);
      // softmax(S / sqrt(d)) with optional mask
      softmax_rows<<<N, blk, 0, stream>>>(scores, scores, N, N, 0.125f, maskMode, cid, counts);
      // O_h = P @ V_h (transposed-B manual kernel)
      wmma_gemm<<<grd(N, D), blk, 0, stream>>>(scores, N, qkv + 2 * H + h * D, H3,
                                               nullptr, attno + h * D, H, N, D, N, 1, 0);
    }
    gemmW(attno, H, wo, H, bo, yout, H, N, H, H, 0);
  };

  // ---- clustering module ----
  gemmW(obs, O, c_fc1_w, O, c_fc1_b, x1, H, N, H, O, 1);
  mha(x1, c_wi, c_bi, c_wo, c_bo, hbuf, 0);
  concat2<<<(2 * N * H + 255) / 256, blk, 0, stream>>>(x1, hbuf, cat2, N, H);
  gemmW(cat2, 2 * H, c_fc2_w, 2 * H, c_fc2_b, zbuf, H, N, H, 2 * H, 1);
  gemmW(zbuf, H, c_head_w, H, c_head_b, cmat, N, N, N, H, 0);
  softmax_rows<<<N, blk, 0, stream>>>(cmat, cmat, N, N, 1.0f, 0, cid, counts);

  // ---- communicate ----
  gemmW(obs, O, t_local_w, O, t_local_b, local, H, N, H, O, 1);
  gemmW(local, H, set_w, H, set_b, sc, 1, N, 1, H, 0);
  seg_init <<<(N + 255) / 256, blk, 0, stream>>>(counts, mkey, wsum, N);
  seg_pass1<<<(N + 255) / 256, blk, 0, stream>>>(cid, sc, counts, mkey, N);
  seg_pass2<<<(N + 255) / 256, blk, 0, stream>>>(cid, sc, mkey, wv, wsum, N);
  fzero     <<<(N * H + 255) / 256, blk, 0, stream>>>(x1, N * H);
  reps_accum<<<(N * H + 255) / 256, blk, 0, stream>>>(cid, wv, local, x1, N, H);
  reps_norm <<<(N * H + 255) / 256, blk, 0, stream>>>(x1, wsum, N, H);
  mha(local, ia_wi, ia_bi, ia_wo, ia_bo, hbuf, 1);   // intra (same-cluster mask)
  mha(x1, ie_wi, ie_bi, ie_wo, ie_bo, zbuf, 2);      // inter over reps (key mask)
  build_after<<<(N * H + 255) / 256, blk, 0, stream>>>(local, zbuf, hbuf, cid, counts, after, N, H);

  // ---- AgentAC head ----
  float* ah1 = cat2;
  float* ah2 = cat2 + (size_t)N * H;
  gemmW(after, 3 * H, a_fc1_w, 3 * H, a_fc1_b, ah1, H, N, H, 3 * H, 1);
  gemmW(ah1, H, a_fc2_w, H, a_fc2_b, ah2, H, N, H, H, 1);
  gemmW(ah2, H, actor_w, H, actor_b, alog, A, N, A, H, 0);
  softmax_rows<<<N, blk, 0, stream>>>(alog, alog, A, A, 1.0f, 3, cid, counts);
  gemmW(ah2, H, value_w, H, value_b, vout, 1, N, 1, H, 0);
}
